// SSA_SC_v7_b3_85615878079114
// MI455X (gfx1250) — matrix operand order repaired and checked by reading, not compiled, not run on hardware
//
#include <hip/hip_runtime.h>
#include <hip/hip_bf16.h>

// ---------------------------------------------------------------------------
// SSA_SC pipeline for gfx1250 (MI455X), wave32 + WMMA f16 16x16x32.
//
// Grid constants (match reference):
//   N=500000 points, CIN=9, MLP 9->64->128->256->128 (BN folded), DPT=128
//   NV2=50000 BEV cells, NV3=200000 voxels, CC=32, B=2, X=Z=256, Y=32
// ---------------------------------------------------------------------------

typedef __attribute__((ext_vector_type(16))) _Float16 v16h;
typedef __attribute__((ext_vector_type(8)))  float    v8f;

#define NPTS   500000
#define NV2C   50000
#define NV3C   200000
#define BDIM   2
#define XDIM   256
#define ZDIM   256
#define YDIM   32
#define CCH    32
#define BN_EPS 1e-5f

// packed-weight (f16) layout offsets, in halfs. Each WMMA B fragment is
// 32 lanes x 16 halfs = 512 halfs, fragment index = kt*NT + nt.
#define OFF_W1 0        //  4 frags (KT=1, NT=4)   9(->32) x 64
#define OFF_W2 2048     // 16 frags (KT=2, NT=8)   64 x 128
#define OFF_W3 10240    // 64 frags (KT=4, NT=16)  128 x 256
#define OFF_W4 43008    // 64 frags (KT=8, NT=8)   256 x 128
#define OFF_WC 75776    //  8 frags (KT=4, NT=2)   128 x 32
#define OFF_WV 79872    //  8 frags (KT=4, NT=2)   128 x 32
#define HALFS_TOTAL 83968

// fused scalar table offsets (floats): s0,t0 (BN0 fold), fused biases
#define OFF_S0 0
#define OFF_T0 16
#define OFF_B1 32
#define OFF_B2 96
#define OFF_B3 224
#define OFF_B4 480
#define OFF_BC 608
#define OFF_BV 640

struct ParamPtrs {
  const float *bn0b,*bn0g,*bn0m,*bn0v;
  const float *bn1b,*bn1g,*bn1m,*bn1v;
  const float *bn2b,*bn2g,*bn2m,*bn2v;
  const float *bn3b,*bn3g,*bn3m,*bn3v;
  const float *Wc,*Bc;
  const float *W1,*B1,*W2,*B2,*W3,*B3,*W4,*B4;
  const float *Wv,*Bv;
};

// ---------------------------------------------------------------------------
// float atomic max via int/uint punning (init value must be -inf bits)
// ---------------------------------------------------------------------------
__device__ __forceinline__ void atomicMaxF(float* addr, float val) {
  if (val >= 0.0f) atomicMax((int*)addr, __float_as_int(val));
  else             atomicMin((unsigned int*)addr, __float_as_uint(val));
}

// ---------------------------------------------------------------------------
// WMMA fragment loaders (CDNA5 16-bit layouts, CDNA5 ISA §7.12.2)
// A (16x32, MxK): lane L holds row M=L&15; halfs[0..7]=K(kb..kb+7),
//   halfs[8..15]=K(kb+16..kb+23), kb = (L<16)?0:8.
// B (32x16, KxN): lane L holds col N=L&15; same K order as A (kb=(L<16)?0:8).
//   Pre-packed so each lane's 16 halfs are contiguous (32B) in memory.
// C/D (16x16 f32): lane L col N=L&15; vgpr r row M=((L<16)?0:8)+r.
// ---------------------------------------------------------------------------
__device__ __forceinline__ v16h load_a_lds(const _Float16* in, int stride,
                                           int kt, int lane) {
  const int m  = lane & 15;
  const int kb = (lane < 16) ? 0 : 8;
  const _Float16* p = in + m * stride + kt * 32 + kb;
  v16h a;
#pragma unroll
  for (int i = 0; i < 8; ++i) { a[i] = p[i]; a[8 + i] = p[16 + i]; }
  return a;
}

__device__ __forceinline__ v16h load_b_packed(const _Float16* wp, int frag,
                                              int lane) {
  return *(const v16h*)(wp + ((size_t)(frag * 32 + lane)) * 16);
}

// one LDS->LDS WMMA layer: 16 rows (one wave) x (KT*32) -> 16 x (NT*16)
template <int KT, int NT, bool RELU>
__device__ __forceinline__ void layer_lds(const _Float16* __restrict__ in,
                                          _Float16* __restrict__ out,
                                          const _Float16* __restrict__ wpack,
                                          const float* __restrict__ bias,
                                          int lane) {
#pragma unroll
  for (int nt = 0; nt < NT; ++nt) {
    v8f c = {};
#pragma unroll
    for (int kt = 0; kt < KT; ++kt) {
      v16h a = load_a_lds(in, KT * 32, kt, lane);
      v16h b = load_b_packed(wpack, kt * NT + nt, lane);
      c = __builtin_amdgcn_wmma_f32_16x16x32_f16(false, a, false, b,
                                                 (short)0, c, false, false);
    }
    const int   n  = lane & 15;
    const int   r0 = (lane < 16) ? 0 : 8;
    const float bb = bias[nt * 16 + n];
#pragma unroll
    for (int r = 0; r < 8; ++r) {
      float v = c[r] + bb;
      if (RELU) v = fmaxf(v, 0.0f);
      out[(r0 + r) * (NT * 16) + nt * 16 + n] = (_Float16)v;
    }
  }
}

// ---------------------------------------------------------------------------
// prep: fold BN into biases + BN0 affine table
// ---------------------------------------------------------------------------
__global__ void prep_scalars_kernel(ParamPtrs P, float* __restrict__ fp) {
  const int t = threadIdx.x;
  if (t < 16) {
    float s = 0.0f, tt = 0.0f;
    if (t < 9) {
      s  = P.bn0g[t] * rsqrtf(P.bn0v[t] + BN_EPS);
      tt = P.bn0b[t] - P.bn0m[t] * s;
    }
    fp[OFF_S0 + t] = s;
    fp[OFF_T0 + t] = tt;
  }
  if (t < 64) {
    float s = P.bn1g[t] * rsqrtf(P.bn1v[t] + BN_EPS);
    fp[OFF_B1 + t] = (P.B1[t] - P.bn1m[t]) * s + P.bn1b[t];
  }
  if (t < 128) {
    float s = P.bn2g[t] * rsqrtf(P.bn2v[t] + BN_EPS);
    fp[OFF_B2 + t] = (P.B2[t] - P.bn2m[t]) * s + P.bn2b[t];
  }
  if (t < 256) {
    float s = P.bn3g[t] * rsqrtf(P.bn3v[t] + BN_EPS);
    fp[OFF_B3 + t] = (P.B3[t] - P.bn3m[t]) * s + P.bn3b[t];
  }
  if (t < 128) fp[OFF_B4 + t] = P.B4[t];
  if (t < 32) { fp[OFF_BC + t] = P.Bc[t]; fp[OFF_BV + t] = P.Bv[t]; }
}

// pack weights (BN scale folded) into f16 B-fragment order. 164 frags total.
__global__ void prep_pack_kernel(ParamPtrs P, _Float16* __restrict__ wp) {
  int f = blockIdx.x;            // 0..163
  const int lane = threadIdx.x;  // 0..31
  const float *W, *g = nullptr, *var = nullptr;
  int K, Nout, NT; size_t off;
  if (f < 4)        { W=P.W1; K=9;   Nout=64;  NT=4;  off=OFF_W1; g=P.bn1g; var=P.bn1v; }
  else if (f < 20)  { W=P.W2; K=64;  Nout=128; NT=8;  off=OFF_W2; g=P.bn2g; var=P.bn2v; f-=4;  }
  else if (f < 84)  { W=P.W3; K=128; Nout=256; NT=16; off=OFF_W3; g=P.bn3g; var=P.bn3v; f-=20; }
  else if (f < 148) { W=P.W4; K=256; Nout=128; NT=8;  off=OFF_W4; f-=84;  }
  else if (f < 156) { W=P.Wc; K=128; Nout=32;  NT=2;  off=OFF_WC; f-=148; }
  else              { W=P.Wv; K=128; Nout=32;  NT=2;  off=OFF_WV; f-=156; }
  const int kt = f / NT, nt = f % NT;
  const int n  = nt * 16 + (lane & 15);
  const int k0 = kt * 32 + ((lane < 16) ? 0 : 8);
  float sc = 1.0f;
  if (g) sc = g[n] * rsqrtf(var[n] + BN_EPS);
  _Float16* dst = wp + off + ((size_t)(f * 32 + lane)) * 16;
#pragma unroll
  for (int i = 0; i < 16; ++i) {
    const int k = k0 + ((i < 8) ? i : 8 + i);
    dst[i] = (_Float16)((k < K) ? W[(size_t)k * Nout + n] * sc : 0.0f);
  }
}

__global__ void fill_kernel(float* __restrict__ p, long long n, float val) {
  long long i = (long long)blockIdx.x * blockDim.x + threadIdx.x;
  if (i < n) p[i] = val;
}

// ---------------------------------------------------------------------------
// fused MLP: 256 threads = 8 waves, each wave owns a 16-row M-tile
// (128 points/block). BN0 applied on load; layers 1-3 WMMA through LDS;
// layer 4 accumulators scattered via atomic float-max into both segment pools.
// ---------------------------------------------------------------------------
__global__ void __launch_bounds__(256, 1)
mlp_kernel(const float* __restrict__ pt_fea,
           const _Float16* __restrict__ wp, const float* __restrict__ fp,
           float* __restrict__ pooled, float* __restrict__ vox,
           const int* __restrict__ unq_inv,
           const int* __restrict__ unq_inv_xyz, int n) {
  extern __shared__ _Float16 smem[];
  _Float16* bufA = smem;                 // [128][256] = 64KB
  _Float16* bufB = smem + 128 * 256;     // [128][128] = 32KB
  const int tid  = threadIdx.x;
  const int lane = tid & 31;
  const int wave = tid >> 5;
  const int base = blockIdx.x * 128;
  const float* s0 = fp + OFF_S0;
  const float* t0 = fp + OFF_T0;

  // stage input -> bufB[128][32] (K padded 9->32), BN0 fused
  for (int idx = tid; idx < 128 * 32; idx += 256) {
    const int row = idx >> 5, k = idx & 31;
    const int g = base + row;
    float v = 0.0f;
    if (k < 9 && g < n) v = pt_fea[(size_t)g * 9 + k] * s0[k] + t0[k];
    bufB[idx] = (_Float16)v;
  }
  __syncthreads();

  const int rb = wave * 16;
  layer_lds<1, 4, true>(bufB + rb * 32,  bufA + rb * 64,  wp + OFF_W1, fp + OFF_B1, lane);
  __syncthreads();
  layer_lds<2, 8, true>(bufA + rb * 64,  bufB + rb * 128, wp + OFF_W2, fp + OFF_B2, lane);
  __syncthreads();
  layer_lds<4, 16, true>(bufB + rb * 128, bufA + rb * 256, wp + OFF_W3, fp + OFF_B3, lane);
  __syncthreads();

  // layer 4: [16][256] x [256][128] -> scatter-max (no LDS epilogue)
  {
    const _Float16* in = bufA + rb * 256;
    const float*    b4 = fp + OFF_B4;
#pragma unroll
    for (int nt = 0; nt < 8; ++nt) {
      v8f c = {};
#pragma unroll
      for (int kt = 0; kt < 8; ++kt) {
        v16h a = load_a_lds(in, 256, kt, lane);
        v16h b = load_b_packed(wp + OFF_W4, kt * 8 + nt, lane);
        c = __builtin_amdgcn_wmma_f32_16x16x32_f16(false, a, false, b,
                                                   (short)0, c, false, false);
      }
      const int   ncol = nt * 16 + (lane & 15);
      const int   r0   = (lane < 16) ? 0 : 8;
      const float bb   = b4[ncol];
#pragma unroll
      for (int r = 0; r < 8; ++r) {
        const int g = base + rb + r0 + r;
        if (g < n) {
          const float v = c[r] + bb;
          atomicMaxF(pooled + (size_t)unq_inv[g] * 128 + ncol, v);
          atomicMaxF(vox + (size_t)unq_inv_xyz[g] * 128 + ncol, v);
        }
      }
    }
  }
}

// ---------------------------------------------------------------------------
// compression GEMM: [rows][128] (f32, -inf sentinel -> 0) x [128][32] + ReLU
// each wave = 16 rows, A fragments converted f32->f16 straight from global.
// ---------------------------------------------------------------------------
__global__ void __launch_bounds__(256)
comp_kernel(const float* __restrict__ seg, int rows,
            const _Float16* __restrict__ wpack, const float* __restrict__ bias,
            float* __restrict__ outp) {
  const int tid  = threadIdx.x;
  const int lane = tid & 31;
  const int wave = tid >> 5;
  const int rowbase = blockIdx.x * 128 + wave * 16;
  int grow = rowbase + (lane & 15);
  if (grow >= rows) grow = rows - 1;               // keep EXEC full for WMMA
  const float* src = seg + (size_t)grow * 128;
  const int kb = (lane < 16) ? 0 : 8;
#pragma unroll
  for (int nt = 0; nt < 2; ++nt) {
    v8f c = {};
#pragma unroll
    for (int kt = 0; kt < 4; ++kt) {
      v16h a;
      const float* p = src + kt * 32 + kb;
#pragma unroll
      for (int i = 0; i < 8; ++i) {
        const float x0 = p[i];
        const float x1 = p[16 + i];
        a[i]     = (_Float16)((x0 > -3.0e38f) ? x0 : 0.0f);  // -inf -> 0
        a[8 + i] = (_Float16)((x1 > -3.0e38f) ? x1 : 0.0f);
      }
      v16h b = load_b_packed(wpack, kt * 2 + nt, lane);
      c = __builtin_amdgcn_wmma_f32_16x16x32_f16(false, a, false, b,
                                                 (short)0, c, false, false);
    }
    const int   ncol = nt * 16 + (lane & 15);
    const int   r0   = (lane < 16) ? 0 : 8;
    const float bb   = bias[ncol];
#pragma unroll
    for (int r = 0; r < 8; ++r) {
      const int rr = rowbase + r0 + r;
      if (rr < rows) outp[(size_t)rr * 32 + ncol] = fmaxf(c[r] + bb, 0.0f);
    }
  }
}

// scatter pp[NV2][32] into dense[B][X][Z][32]
__global__ void scatter_kernel(const float* __restrict__ pp,
                               const int* __restrict__ unq,
                               float* __restrict__ dense, int nv2) {
  const int t = blockIdx.x * blockDim.x + threadIdx.x;
  if (t >= nv2 * 32) return;
  const int i = t >> 5, c = t & 31;
  const int b = unq[i * 3 + 0], x = unq[i * 3 + 1], z = unq[i * 3 + 2];
  dense[((((size_t)b * XDIM + x) * ZDIM + z) * CCH) + c] = pp[(size_t)i * 32 + c];
}

// out[b][ch][x][z]: ch<32 -> occu[b][ch][x][z]; else 3x3 maxpool of dense[...,ch-32]
__global__ void final_kernel(const float* __restrict__ occu,
                             const float* __restrict__ dense,
                             float* __restrict__ outp) {
  const long long t = (long long)blockIdx.x * blockDim.x + threadIdx.x;
  if (t >= (long long)BDIM * 64 * XDIM * ZDIM) return;
  const int z  = (int)(t & 255);
  const int x  = (int)((t >> 8) & 255);
  const int ch = (int)((t >> 16) & 63);
  const int b  = (int)(t >> 22);
  float v;
  if (ch < YDIM) {
    v = occu[(((size_t)b * YDIM + ch) * XDIM + x) * ZDIM + z];
  } else {
    const int c = ch - YDIM;
    v = -INFINITY;
#pragma unroll
    for (int dx = -1; dx <= 1; ++dx) {
      const int xx = x + dx;
      if (xx < 0 || xx >= XDIM) continue;
#pragma unroll
      for (int dz = -1; dz <= 1; ++dz) {
        const int zz = z + dz;
        if (zz < 0 || zz >= ZDIM) continue;
        v = fmaxf(v, dense[((((size_t)b * XDIM + xx) * ZDIM + zz) * CCH) + c]);
      }
    }
  }
  outp[t] = v;
}

// ---------------------------------------------------------------------------
// launch
// ---------------------------------------------------------------------------
extern "C" void kernel_launch(void* const* d_in, const int* in_sizes, int n_in,
                              void* d_out, int out_size, void* d_ws,
                              size_t ws_size, hipStream_t stream) {
  (void)in_sizes; (void)n_in; (void)out_size; (void)ws_size;
  const float* pt_fea      = (const float*)d_in[0];
  const float* occu        = (const float*)d_in[1];
  const int*   unq_inv     = (const int*)d_in[2];
  const int*   unq_inv_xyz = (const int*)d_in[3];
  const int*   unq         = (const int*)d_in[4];
  // d_in[5] = unq_xyz (unused: pv is returned flat, not scattered)
  // params leaves, jax pytree order (sorted dict keys):
  // bn0..bn3 {beta,gamma,mean,var}, comp{W,b}, l1..l4{W,b}, vcomp{W,b}
  ParamPtrs P;
  int p = 6;
  P.bn0b=(const float*)d_in[p+0];  P.bn0g=(const float*)d_in[p+1];
  P.bn0m=(const float*)d_in[p+2];  P.bn0v=(const float*)d_in[p+3];
  P.bn1b=(const float*)d_in[p+4];  P.bn1g=(const float*)d_in[p+5];
  P.bn1m=(const float*)d_in[p+6];  P.bn1v=(const float*)d_in[p+7];
  P.bn2b=(const float*)d_in[p+8];  P.bn2g=(const float*)d_in[p+9];
  P.bn2m=(const float*)d_in[p+10]; P.bn2v=(const float*)d_in[p+11];
  P.bn3b=(const float*)d_in[p+12]; P.bn3g=(const float*)d_in[p+13];
  P.bn3m=(const float*)d_in[p+14]; P.bn3v=(const float*)d_in[p+15];
  P.Wc=(const float*)d_in[p+16];   P.Bc=(const float*)d_in[p+17];
  P.W1=(const float*)d_in[p+18];   P.B1=(const float*)d_in[p+19];
  P.W2=(const float*)d_in[p+20];   P.B2=(const float*)d_in[p+21];
  P.W3=(const float*)d_in[p+22];   P.B3=(const float*)d_in[p+23];
  P.W4=(const float*)d_in[p+24];   P.B4=(const float*)d_in[p+25];
  P.Wv=(const float*)d_in[p+26];   P.Bv=(const float*)d_in[p+27];

  // workspace layout
  char*     ws = (char*)d_ws;
  _Float16* wp = (_Float16*)ws;                       // 83968 halfs = 167936 B
  float*    fp = (float*)(ws + HALFS_TOTAL * 2);      // 672 floats (pad 4KB)
  float* pooled = (float*)(ws + HALFS_TOTAL * 2 + 4096);
  float* vox    = pooled + (size_t)NV2C * 128;        // 6.4M floats
  float* pp     = vox + (size_t)NV3C * 128;           // +25.6M floats
  float* dense  = pp + (size_t)NV2C * 32;             // +1.6M floats
                                                      // dense: 4.19M floats
  float* out_main = (float*)d_out;                    // [2][64][256][256]
  float* out_pv   = out_main + (size_t)BDIM * 64 * XDIM * ZDIM;  // [NV3][32]

  // 1) fold BN + pack WMMA weight fragments
  prep_scalars_kernel<<<1, 256, 0, stream>>>(P, fp);
  prep_pack_kernel<<<164, 32, 0, stream>>>(P, wp);

  // 2) init segment-max targets to -inf, dense grid to 0
  {
    const long long nseg = (long long)NV2C * 128 + (long long)NV3C * 128;
    fill_kernel<<<(unsigned)((nseg + 255) / 256), 256, 0, stream>>>(pooled, nseg, -INFINITY);
    const long long nd = (long long)BDIM * XDIM * ZDIM * CCH;
    fill_kernel<<<(unsigned)((nd + 255) / 256), 256, 0, stream>>>(dense, nd, 0.0f);
  }

  // 3) fused WMMA MLP + dual scatter-max (128 points / block)
  mlp_kernel<<<(NPTS + 127) / 128, 256, (128 * 256 + 128 * 128) * sizeof(_Float16),
               stream>>>(pt_fea, wp, fp, pooled, vox, unq_inv, unq_inv_xyz, NPTS);

  // 4) compression GEMMs (WMMA): pooled->pp, vox->pv (pv straight to d_out)
  comp_kernel<<<(NV2C + 127) / 128, 256, 0, stream>>>(pooled, NV2C, wp + OFF_WC,
                                                      fp + OFF_BC, pp);
  comp_kernel<<<(NV3C + 127) / 128, 256, 0, stream>>>(vox, NV3C, wp + OFF_WV,
                                                      fp + OFF_BV, out_pv);

  // 5) dense scatter + maxpool/concat epilogue
  scatter_kernel<<<(NV2C * 32 + 255) / 256, 256, 0, stream>>>(pp, unq, dense, NV2C);
  final_kernel<<<(BDIM * 64 * XDIM * ZDIM + 255) / 256, 256, 0, stream>>>(
      occu, dense, out_main);
}
